// behavior_model_87686052315338
// MI455X (gfx1250) — matrix operand order repaired and checked by reading, not compiled, not run on hardware
//
#include <hip/hip_runtime.h>
#include <hip/hip_bf16.h>

#define B_   64
#define N_   16
#define P_   20
#define NB_  30
#define C_   1024
#define F_   14
#define NIMG (B_*17)      // 1088 roi images (1 global + 16 persons per batch)
#define FEAT 2304         // 256*3*3

typedef __bf16 bf16_t;
typedef bf16_t v16bf __attribute__((ext_vector_type(16)));
typedef float  v8f   __attribute__((ext_vector_type(8)));

union Frag { v16bf v; unsigned int w[8]; unsigned short u[16]; };

__device__ __forceinline__ unsigned short f2bf(float f){
  unsigned int x = __float_as_uint(f);
  return (unsigned short)((x + 0x7FFFu + ((x >> 16) & 1u)) >> 16); // RNE
}
__device__ __forceinline__ float bf2f(unsigned short u){
  return __uint_as_float(((unsigned int)u) << 16);
}
__device__ __forceinline__ v8f wmma_bf(const Frag& a, const Frag& b, v8f c){
  return __builtin_amdgcn_wmma_f32_16x16x32_bf16(false, a.v, false, b.v,
                                                 (short)0, c, false, false);
}
__device__ __forceinline__ v8f vzero8(){ v8f z = {0.f,0.f,0.f,0.f,0.f,0.f,0.f,0.f}; return z; }

// ---------------- precompute kernels ----------------
__global__ void k_bnprep(const float* g1,const float* b1,const float* m1,const float* v1,
                         const float* g2,const float* b2,const float* m2,const float* v2,
                         float* s1,float* t1,float* s2,float* t2){
  int i = blockIdx.x*blockDim.x + threadIdx.x;
  if(i < 512){ float s = g1[i]*rsqrtf(v1[i]+1e-5f); s1[i]=s; t1[i]=b1[i]-m1[i]*s; }
  if(i < 256){ float s = g2[i]*rsqrtf(v2[i]+1e-5f); s2[i]=s; t2[i]=b2[i]-m2[i]*s; }
}

__global__ void k_zero16(unsigned short* p, int n){
  int i = blockIdx.x*blockDim.x + threadIdx.x;
  if(i < n) p[i] = 0;
}

// conv w [OC][CIN][3][3] f32 -> bf16 [OC][9][CIN]  (tap-major K, ic contiguous)
__global__ void k_cvt_convw(const float* in, unsigned short* out, int OC, int CIN){
  int i = blockIdx.x*blockDim.x + threadIdx.x;
  if(i >= OC*CIN*9) return;
  int ic = i % CIN; int r = i / CIN; int t = r % 9; int oc = r / 9;
  out[i] = f2bf(in[(oc*CIN + ic)*9 + t]);
}
// conv1d_w [2304][2304][3] -> bf16 [3][2304][2304]  (tap-major, ic contiguous)
__global__ void k_cvt_w1d(const float* in, unsigned short* out){
  int i = blockIdx.x*blockDim.x + threadIdx.x;
  if(i >= 3*2304*2304) return;
  int ic = i % 2304; int r = i / 2304; int oc = r % 2304; int t = r / 2304;
  out[i] = f2bf(in[(oc*2304 + ic)*3 + t]);
}
// fc1_w [2304][1024] -> bf16 [1024][2304] (transposed, k contiguous)
__global__ void k_cvt_fc1t(const float* in, unsigned short* out){
  int i = blockIdx.x*blockDim.x + threadIdx.x;
  if(i >= 1024*2304) return;
  int k = i % 2304; int n = i / 2304;
  out[i] = f2bf(in[k*1024 + n]);
}
// fc2_w [1024][30] -> bf16 [32][1024] (transposed, zero-padded columns 30..31)
__global__ void k_cvt_fc2t(const float* in, unsigned short* out){
  int i = blockIdx.x*blockDim.x + threadIdx.x;
  if(i >= 32*1024) return;
  int k = i % 1024; int n = i / 1024;
  out[i] = (n < 30) ? f2bf(in[k*30 + n]) : (unsigned short)0;
}

// ---------------- ROI align ----------------
// out roi_pm [NIMG][9][C] bf16 (position-major) ; img = b*17 + j (j==0: global box)
__global__ void k_roi(const float* __restrict__ fmap, const float* __restrict__ boxes,
                      unsigned short* __restrict__ roi_pm){
  long long gid = (long long)blockIdx.x*blockDim.x + threadIdx.x;
  const long long total = (long long)NIMG*9*C_;
  if(gid >= total) return;
  int c   = (int)(gid % C_);          // fastest: coalesced store
  long long r = gid / C_;
  int p   = (int)(r % 9);
  int img = (int)(r / 9);
  int b = img/17, j = img%17;
  float x1,y1,x2,y2;
  if(j==0){ x1=0.f; y1=0.f; x2=(float)F_; y2=(float)F_; }
  else{ const float* bx = boxes + ((b*16)+(j-1))*4; x1=bx[0]; y1=bx[1]; x2=bx[2]; y2=bx[3]; }
  float rw = fmaxf(x2-x1, 1.f), rh = fmaxf(y2-y1, 1.f);
  int oy = p/3, ox = p%3;
  const float* fm = fmap + ((long long)b*C_ + c)*(F_*F_);
  float acc = 0.f;
  #pragma unroll
  for(int sy=0; sy<2; sy++){
    float ys = y1 + ((float)(oy*2+sy)+0.5f)*0.5f*(rh*(1.f/3.f));
    ys = fminf(fmaxf(ys, 0.f), 13.f);
    float y0 = floorf(ys); int y0i=(int)y0; int y1i = min(y0i+1, 13); float wy = ys - y0;
    #pragma unroll
    for(int sx=0; sx<2; sx++){
      float xs = x1 + ((float)(ox*2+sx)+0.5f)*0.5f*(rw*(1.f/3.f));
      xs = fminf(fmaxf(xs, 0.f), 13.f);
      float x0 = floorf(xs); int x0i=(int)x0; int x1i = min(x0i+1, 13); float wx = xs - x0;
      float v00 = fm[y0i*14 + x0i], v01 = fm[y0i*14 + x1i];
      float v10 = fm[y1i*14 + x0i], v11 = fm[y1i*14 + x1i];
      acc += v00*(1.f-wy)*(1.f-wx) + v01*(1.f-wy)*wx + v10*wy*(1.f-wx) + v11*wy*wx;
    }
  }
  roi_pm[gid] = f2bf(acc*0.25f);
}

// ---------------- conv3x3 (SAME) on 3x3 spatial as implicit GEMM ----------------
// xin [img][9][CIN] bf16 (position-major) ; wbf [COUT][9][CIN] bf16 (tap-major K)
// OUT_PM=true : yout [img][9][COUT] ; OUT_PM=false : yout [img][COUT*9] (f = oc*9+p)
template<int CIN, int COUT, int MTW, bool OUT_PM>
__global__ void __launch_bounds__(256)
k_conv(const unsigned short* __restrict__ xin, const unsigned short* __restrict__ wbf,
       const float* __restrict__ bscale, const float* __restrict__ bbias,
       unsigned short* __restrict__ yout){
  __shared__ unsigned short xpad[26*CIN];   // positions 0..24 = padded 5x5, 25 = zeros
  const int img = blockIdx.x;
  const int tid = threadIdx.x;
  for(int i=tid; i<26*CIN; i+=256){
    int pos = i/CIN, c = i - pos*CIN;
    int y = pos/5, x = pos - y*5;
    unsigned short v = 0;
    if(pos < 25 && y>=1 && y<=3 && x>=1 && x<=3)
      v = xin[((long long)img*9 + (y-1)*3 + (x-1))*CIN + c];
    xpad[i] = v;
  }
  __syncthreads();
  const int wave = tid>>5, lane = tid&31;
  const int col = lane&15, hi = lane>>4;
  const int py = col/3, px = col - py*3;     // only meaningful for col<9
  v8f acc[MTW];
  #pragma unroll
  for(int m=0;m<MTW;m++) acc[m] = vzero8();

  #pragma unroll
  for(int t=0;t<9;t++){
    const int ty = t/3, tx = t - ty*3;
    const int pos = (col < 9) ? ((py+ty)*5 + (px+tx)) : 25;  // 25 -> zero slot
    const unsigned short* bsrc = xpad + pos*CIN;
    for(int kb=0; kb<CIN; kb+=32){
      Frag bf;
      int klo = kb + (hi<<3);
      #pragma unroll
      for(int i=0;i<8;i++) bf.w[i] = *(const unsigned int*)(bsrc + klo + (i<4 ? 2*i : 16 + 2*(i-4)));
      int kloa = kb + (hi ? 8 : 0);
      Frag af[MTW];
      #pragma unroll
      for(int m=0;m<MTW;m++){
        int row = (wave*MTW + m)*16 + col;
        const unsigned short* abase = wbf + ((long long)row*9 + t)*CIN + kloa;
        #pragma unroll
        for(int i=0;i<4;i++){
          af[m].w[i]   = *(const unsigned int*)(abase + 2*i);
          af[m].w[4+i] = *(const unsigned int*)(abase + 16 + 2*i);
        }
      }
      #pragma unroll
      for(int m=0;m<MTW;m++) acc[m] = wmma_bf(af[m], bf, acc[m]);
    }
  }
  if(col < 9){
    #pragma unroll
    for(int m=0;m<MTW;m++){
      #pragma unroll
      for(int i=0;i<8;i+=2){
        int oc = (wave*MTW + m)*16 + i + 8*hi;
        float v0 = acc[m][i]  *bscale[oc]   + bbias[oc];
        float v1 = acc[m][i+1]*bscale[oc+1] + bbias[oc+1];
        v0 = v0 > 0.f ? v0 : 0.1f*v0;
        v1 = v1 > 0.f ? v1 : 0.1f*v1;
        if(OUT_PM){
          unsigned int pk = (unsigned int)f2bf(v0) | ((unsigned int)f2bf(v1) << 16);
          *(unsigned int*)(yout + ((long long)img*9 + col)*COUT + oc) = pk;
        } else {
          yout[(long long)img*COUT*9 + (oc  )*9 + col] = f2bf(v0);
          yout[(long long)img*COUT*9 + (oc+1)*9 + col] = f2bf(v1);
        }
      }
    }
  }
}

// ---------------- behavior assembly (into zero-padded [B+2][P][2304]) ----------------
__global__ void k_assemble(const unsigned short* __restrict__ y2, const int* __restrict__ pidx,
                           unsigned short* __restrict__ behp){
  int i = blockIdx.x*blockDim.x + threadIdx.x;
  if(i >= B_*P_*FEAT) return;
  int f = i % FEAT; int r = i / FEAT; int p = r % P_; int b = r / P_;
  float val = bf2f(y2[((long long)(b*17))*FEAT + f]);          // global roi slot
  #pragma unroll
  for(int n=0;n<16;n++){
    if(pidx[b*16+n] == p) val += bf2f(y2[((long long)(b*17+1+n))*FEAT + f]);
  }
  behp[((long long)(b+1)*P_ + p)*FEAT + f] = f2bf(val);
}

// ---------------- conv1d over batch axis (k=3, SAME), per person-slot p ----------------
// behp [B+2][P][2304] bf16 (rows 0 and B+1 zero) ; w [3][2304][2304] bf16
// xfeat [B][P][2304] bf16 (bias fused)
// B-tile (66 padded batch rows x 32 K) staged to LDS once per K-step with
// global_load_async_to_lds_b32 and shared by all 8 waves (12 WMMAs each).
__global__ void __launch_bounds__(256)
k_conv1d(const unsigned short* __restrict__ behp, const unsigned short* __restrict__ w,
         const float* __restrict__ bias, unsigned short* __restrict__ xfeat){
  __shared__ unsigned short btile[66*32];        // [row 0..65][k 0..31]
  const int p = blockIdx.y;
  const int tid = threadIdx.x, wave = tid>>5, lane = tid&31;
  const int col = lane&15, hi = lane>>4;
  const int mtile = blockIdx.x*8 + wave;         // 0..143  (2304/16)
  v8f acc[4];
  #pragma unroll
  for(int q=0;q<4;q++) acc[q] = vzero8();

  for(int kb=0; kb<2304; kb+=32){
    // async-stage B tile: 66 rows x 16 dwords
    for(int u=tid; u<66*16; u+=256){
      int row = u >> 4, dw = u & 15;
      const unsigned int* g =
        (const unsigned int*)(behp + ((long long)row*P_ + p)*FEAT + kb) + dw;
      unsigned ldso = (unsigned)(unsigned long long)&btile[row*32 + dw*2];
      asm volatile("global_load_async_to_lds_b32 %0, %1, off"
                   :: "v"(ldso), "v"(g) : "memory");
    }
    asm volatile("s_wait_asynccnt 0" ::: "memory");
    __syncthreads();

    #pragma unroll
    for(int t=0;t<3;t++){
      const unsigned short* abase =
        w + ((long long)t*2304 + mtile*16 + col)*2304 + kb + (hi ? 8 : 0);
      Frag af;
      #pragma unroll
      for(int i=0;i<4;i++){
        af.w[i]   = *(const unsigned int*)(abase + 2*i);
        af.w[4+i] = *(const unsigned int*)(abase + 16 + 2*i);
      }
      Frag bfr[4];
      #pragma unroll
      for(int nt=0;nt<4;nt++){
        int row = nt*16 + col + t;               // padded row index, always in range
        const unsigned short* bb = btile + row*32 + (hi<<3);
        #pragma unroll
        for(int i=0;i<8;i++) bfr[nt].w[i] = *(const unsigned int*)(bb + (i<4 ? 2*i : 16 + 2*(i-4)));
      }
      #pragma unroll
      for(int nt=0;nt<4;nt++) acc[nt] = wmma_bf(af, bfr[nt], acc[nt]);
    }
    __syncthreads();
  }
  #pragma unroll
  for(int nt=0;nt<4;nt++){
    int bcol = nt*16 + col;
    #pragma unroll
    for(int i=0;i<8;i++){
      int oc = mtile*16 + i + 8*hi;
      xfeat[((long long)bcol*P_ + p)*FEAT + oc] = f2bf(acc[nt][i] + bias[oc]);
    }
  }
}

// ---------------- fc1: gather rows via person_idx, GEMM, relu ----------------
__global__ void __launch_bounds__(256)
k_fc1(const unsigned short* __restrict__ xfeat, const int* __restrict__ pidx,
      const unsigned short* __restrict__ wt, const float* __restrict__ bias,
      unsigned short* __restrict__ hout){
  const int tid = threadIdx.x, wave = tid>>5, lane = tid&31;
  const int col = lane&15, hi = lane>>4;
  int W = blockIdx.x*8 + wave;                  // 0..511
  int mtile = W>>3;                              // 0..63 == batch b
  int ngrp  = W&7;                               // 8 N-tiles each
  int idx = pidx[mtile*16 + col];
  const unsigned short* abase = xfeat + ((long long)mtile*P_ + idx)*FEAT;
  v8f acc[8];
  #pragma unroll
  for(int q=0;q<8;q++) acc[q] = vzero8();

  for(int kb=0; kb<FEAT; kb+=32){
    int kloa = kb + (hi ? 8 : 0);
    Frag af;
    #pragma unroll
    for(int i=0;i<4;i++){
      af.w[i]   = *(const unsigned int*)(abase + kloa + 2*i);
      af.w[4+i] = *(const unsigned int*)(abase + kloa + 16 + 2*i);
    }
    int klo = kb + (hi<<3);
    #pragma unroll
    for(int half=0; half<2; half++){
      Frag bfr[4];
      #pragma unroll
      for(int q=0;q<4;q++){
        int n = (ngrp*8 + half*4 + q)*16 + col;
        const unsigned short* bb = wt + (long long)n*FEAT + klo;
        #pragma unroll
        for(int i=0;i<8;i++) bfr[q].w[i] = *(const unsigned int*)(bb + (i<4 ? 2*i : 16 + 2*(i-4)));
      }
      #pragma unroll
      for(int q=0;q<4;q++) acc[half*4+q] = wmma_bf(af, bfr[q], acc[half*4+q]);
    }
  }
  #pragma unroll
  for(int nt=0;nt<8;nt++){
    int n = (ngrp*8 + nt)*16 + col;
    #pragma unroll
    for(int i=0;i<8;i++){
      int r = mtile*16 + i + 8*hi;
      float v = acc[nt][i] + bias[n];
      v = v > 0.f ? v : 0.f;                       // relu
      hout[(long long)r*1024 + n] = f2bf(v);
    }
  }
}

// ---------------- fc2: [1024,1024] @ [1024,30] -> d_out f32 ----------------
__global__ void __launch_bounds__(256)
k_fc2(const unsigned short* __restrict__ h, const unsigned short* __restrict__ wt,
      const float* __restrict__ bias, float* __restrict__ out){
  const int tid = threadIdx.x, wave = tid>>5, lane = tid&31;
  const int col = lane&15, hi = lane>>4;
  int mtile = blockIdx.x*8 + wave;               // 0..63
  const unsigned short* abase = h + (long long)(mtile*16 + col)*1024;
  v8f acc[2];
  acc[0] = vzero8(); acc[1] = vzero8();
  for(int kb=0; kb<1024; kb+=32){
    int kloa = kb + (hi ? 8 : 0);
    Frag af;
    #pragma unroll
    for(int i=0;i<4;i++){
      af.w[i]   = *(const unsigned int*)(abase + kloa + 2*i);
      af.w[4+i] = *(const unsigned int*)(abase + kloa + 16 + 2*i);
    }
    int klo = kb + (hi<<3);
    Frag bfr[2];
    #pragma unroll
    for(int nt=0;nt<2;nt++){
      int n = nt*16 + col;
      const unsigned short* bb = wt + (long long)n*1024 + klo;
      #pragma unroll
      for(int i=0;i<8;i++) bfr[nt].w[i] = *(const unsigned int*)(bb + (i<4 ? 2*i : 16 + 2*(i-4)));
    }
    #pragma unroll
    for(int nt=0;nt<2;nt++) acc[nt] = wmma_bf(af, bfr[nt], acc[nt]);
  }
  #pragma unroll
  for(int nt=0;nt<2;nt++){
    int j = nt*16 + col;
    if(j < NB_){
      #pragma unroll
      for(int i=0;i<8;i++){
        int r = mtile*16 + i + 8*hi;
        out[r*NB_ + j] = acc[nt][i] + bias[j];
      }
    }
  }
}

// ---------------- launch ----------------
extern "C" void kernel_launch(void* const* d_in, const int* in_sizes, int n_in,
                              void* d_out, int out_size, void* d_ws, size_t ws_size,
                              hipStream_t stream){
  (void)in_sizes; (void)n_in; (void)out_size; (void)ws_size;
  const float* fmap    = (const float*)d_in[0];
  const float* boxes   = (const float*)d_in[1];
  const int*   pidx    = (const int*)  d_in[2];
  const float* conv1_w = (const float*)d_in[3];
  const float* bn1_g   = (const float*)d_in[4];
  const float* bn1_b   = (const float*)d_in[5];
  const float* bn1_m   = (const float*)d_in[6];
  const float* bn1_v   = (const float*)d_in[7];
  const float* conv2_w = (const float*)d_in[8];
  const float* bn2_g   = (const float*)d_in[9];
  const float* bn2_b   = (const float*)d_in[10];
  const float* bn2_m   = (const float*)d_in[11];
  const float* bn2_v   = (const float*)d_in[12];
  const float* conv1d_w= (const float*)d_in[13];
  const float* conv1d_b= (const float*)d_in[14];
  const float* fc1_w   = (const float*)d_in[15];
  const float* fc1_b   = (const float*)d_in[16];
  const float* fc2_w   = (const float*)d_in[17];
  const float* fc2_b   = (const float*)d_in[18];
  float* out = (float*)d_out;

  char* ws = (char*)d_ws; size_t off = 0;
  auto alloc = [&](size_t bytes)->void*{
    void* p = ws + off; off = (off + bytes + 255) & ~(size_t)255; return p; };

  unsigned short* w1b   = (unsigned short*)alloc((size_t)512*9216*2);
  unsigned short* w2b   = (unsigned short*)alloc((size_t)256*4608*2);
  unsigned short* w1db  = (unsigned short*)alloc((size_t)3*2304*2304*2);
  unsigned short* fc1t  = (unsigned short*)alloc((size_t)1024*2304*2);
  unsigned short* fc2t  = (unsigned short*)alloc((size_t)32*1024*2);
  float* s1 = (float*)alloc(512*4);  float* t1 = (float*)alloc(512*4);
  float* s2 = (float*)alloc(256*4);  float* t2 = (float*)alloc(256*4);
  unsigned short* roib  = (unsigned short*)alloc((size_t)NIMG*9*C_*2);      // position-major
  unsigned short* y1    = (unsigned short*)alloc((size_t)NIMG*9*512*2);     // position-major
  unsigned short* y2    = (unsigned short*)alloc((size_t)NIMG*FEAT*2);      // f = oc*9+p
  unsigned short* behp  = (unsigned short*)alloc((size_t)(B_+2)*P_*FEAT*2); // zero-padded
  unsigned short* xfeat = (unsigned short*)alloc((size_t)B_*P_*FEAT*2);
  unsigned short* hbuf  = (unsigned short*)alloc((size_t)1024*1024*2);

  k_bnprep<<<2,256,0,stream>>>(bn1_g,bn1_b,bn1_m,bn1_v, bn2_g,bn2_b,bn2_m,bn2_v,
                               s1,t1,s2,t2);
  { int n = 512*9216; k_cvt_convw<<<(n+255)/256,256,0,stream>>>(conv1_w, w1b, 512, 1024); }
  { int n = 256*4608; k_cvt_convw<<<(n+255)/256,256,0,stream>>>(conv2_w, w2b, 256, 512); }
  { int n = 3*2304*2304; k_cvt_w1d<<<(n+255)/256,256,0,stream>>>(conv1d_w, w1db); }
  { int n = 1024*2304; k_cvt_fc1t<<<(n+255)/256,256,0,stream>>>(fc1_w, fc1t); }
  { int n = 32*1024;   k_cvt_fc2t<<<(n+255)/256,256,0,stream>>>(fc2_w, fc2t); }
  { int n = (B_+2)*P_*FEAT; k_zero16<<<(n+255)/256,256,0,stream>>>(behp, n); } // pad rows
  { long long n = (long long)NIMG*9*C_;
    k_roi<<<(unsigned)((n+255)/256),256,0,stream>>>(fmap, boxes, roib); }

  k_conv<1024,512,4,true ><<<NIMG,256,0,stream>>>(roib, w1b, s1, t1, y1);
  k_conv< 512,256,2,false><<<NIMG,256,0,stream>>>(y1,   w2b, s2, t2, y2);

  { int n = B_*P_*FEAT; k_assemble<<<(n+255)/256,256,0,stream>>>(y2, pidx, behp); }

  dim3 g1d(18, P_);   // 144 M-tiles / 8 waves, one y-slice per person slot
  k_conv1d<<<g1d,256,0,stream>>>(behp, w1db, conv1d_b, xfeat);

  k_fc1<<<64,256,0,stream>>>(xfeat, pidx, fc1t, fc1_b, hbuf);
  k_fc2<<<8,256,0,stream>>>(hbuf, fc2t, fc2_b, out);
}
